// CGCNN_22170621182513
// MI455X (gfx1250) — matrix operand order repaired and checked by reading, not compiled, not run on hardware
//
#include <hip/hip_runtime.h>
#include <hip/hip_bf16.h>

// ---------------- problem constants ----------------
#define NN   50000
#define EE   800000
#define BB   512
#define HH   128
#define NDIM 91
#define EDIM 20
#define ZDIM 276          // 2*H + ED
#define ZP   288          // padded to multiple of 32
#define KT_E 9            // ZP/32
#define KT_J 16           // 512/32
#define NT   8            // H/16
#define TE   64           // edges per block in k_edge
#define EPSV 1e-5f

typedef __attribute__((ext_vector_type(16))) __bf16 v16bf;
typedef __attribute__((ext_vector_type(8)))  __bf16 v8bf;
typedef __attribute__((ext_vector_type(8)))  float  v8f;

__device__ __forceinline__ unsigned short f2b(float f) {
    unsigned u = __float_as_uint(f);
    unsigned r = (u + 0x7fffu + ((u >> 16) & 1u)) >> 16;   // RNE
    return (unsigned short)r;
}
__device__ __forceinline__ float sigm(float x) { return 1.f / (1.f + __expf(-x)); }
__device__ __forceinline__ unsigned fenc(float f) {
    unsigned u = __float_as_uint(f);
    return (u & 0x80000000u) ? ~u : (u | 0x80000000u);
}
__device__ __forceinline__ float fdec(unsigned u) {
    return __uint_as_float((u & 0x80000000u) ? (u & 0x7fffffffu) : ~u);
}
__device__ __forceinline__ v16bf cat8(v8bf lo, v8bf hi) {
    return __builtin_shufflevector(lo, hi, 0,1,2,3,4,5,6,7,8,9,10,11,12,13,14,15);
}

// ---------------- utility fills ----------------
__global__ void k_fillf(float* p, float v, int n) {
    int i = blockIdx.x * 256 + threadIdx.x;
    if (i < n) p[i] = v;
}
__global__ void k_fillu(unsigned* p, unsigned v, int n) {
    int i = blockIdx.x * 256 + threadIdx.x;
    if (i < n) p[i] = v;
}

// ---------------- bf16 conversions ----------------
__global__ void k_cvt_ef(const float* ef, unsigned short* out, int n) {
    int i = blockIdx.x * 256 + threadIdx.x;
    if (i < n) out[i] = f2b(ef[i]);
}

// Pre-shuffle a (Kreal x 128) f32 weight matrix into WMMA B-fragment-major bf16:
// out[((kt*NT+nt)*512) + lane*16 + i] = W[kt*32 + (i<8 ? (lane>>4)*8 + i : 16 + (lane>>4)*8 + i-8)][nt*16 + (lane&15)]
__global__ void k_format_w(const float* W, unsigned short* out, int Kreal, int KT) {
    int idx = blockIdx.x * 256 + threadIdx.x;
    int total = KT * NT * 512;
    if (idx >= total) return;
    int frag = idx >> 9, within = idx & 511;
    int lane = within >> 4, i = within & 15;
    int kt = frag >> 3, nt = frag & 7;
    int k = kt * 32 + (i < 8 ? (lane >> 4) * 8 + i : 16 + (lane >> 4) * 8 + (i - 8));
    int col = nt * 16 + (lane & 15);
    float v = (k < Kreal) ? W[k * HH + col] : 0.f;
    out[idx] = f2b(v);
}

// ---------------- node embedding: h = nf @ emb_w + emb_b ----------------
__global__ void k_embed(const float* nf, const float* ew, const float* eb,
                        float* h, unsigned short* h_bf, unsigned short* states) {
    int idx = blockIdx.x * 256 + threadIdx.x;
    if (idx >= NN * HH) return;
    int n = idx >> 7, c = idx & 127;
    float acc = eb[c];
    for (int k = 0; k < NDIM; ++k) acc = fmaf(nf[n * NDIM + k], ew[k * HH + c], acc);
    h[idx] = acc;
    unsigned short b = f2b(acc);
    h_bf[idx] = b;
    states[n * 512 + c] = b;
}

// ---------------- in-degree ----------------
__global__ void k_deg(const int* eidx, float* deg) {
    int e = blockIdx.x * 256 + threadIdx.x;
    if (e < EE) atomicAdd(&deg[eidx[EE + e]], 1.0f);
}

// ---------------- fused edge message kernel -----------------------------------------
// block = 256 threads = 8 waves; tile = 64 edges (4 row-tiles of 16).
// Wave w owns N-tile w (16 output channels) for BOTH gate and msg GEMMs and sweeps the
// 4 row-tiles, so every B fragment load feeds 4 WMMAs (4x less L2 weight traffic).
__global__ void __launch_bounds__(256)
k_edge(const unsigned short* __restrict__ h_bf, const unsigned short* __restrict__ ef_bf,
       const int* __restrict__ eidx,
       const unsigned short* __restrict__ wg, const unsigned short* __restrict__ wm,
       const float* __restrict__ gb, const float* __restrict__ mb,
       float* __restrict__ agg) {
    __shared__ alignas(32) unsigned short zt[TE * ZP];   // 36864 B
    __shared__ int srcs[TE];
    __shared__ int dsts[TE];
    const int e0 = blockIdx.x * TE;
    const int tid = threadIdx.x;

    if (tid < TE)            srcs[tid] = eidx[e0 + tid];
    else if (tid < 2 * TE)   dsts[tid - TE] = eidx[EE + e0 + (tid - TE)];
    __syncthreads();

    // stage z rows: [h[src](128) | h[dst](128) | ef(20) | 0-pad(12)] in bf16
    // h parts: 32 chunks of 16B per row, coalesced b128 gather -> ds b128
    for (int c = tid; c < TE * 32; c += 256) {
        int m = c >> 5, ch = c & 31;
        int node = (ch < 16) ? srcs[m] : dsts[m];
        const uint4 v = *reinterpret_cast<const uint4*>(h_bf + node * HH + (ch & 15) * 8);
        *reinterpret_cast<uint4*>(zt + m * ZP + ch * 8) = v;
    }
    // ef + pad: 16 dword chunks per row (10 dwords of ef, 6 dwords of zero)
    for (int c = tid; c < TE * 16; c += 256) {
        int m = c >> 4, j = c & 15;
        unsigned v = (j < 10)
            ? *reinterpret_cast<const unsigned*>(ef_bf + (e0 + m) * EDIM + j * 2) : 0u;
        *reinterpret_cast<unsigned*>(zt + m * ZP + 2 * HH + j * 2) = v;
    }
    __syncthreads();

    const int lane = tid & 31, wave = tid >> 5;    // wave = owned N-tile
    const int row = lane & 15, hi = lane >> 4;
    v8f ag[4] = {{}, {}, {}, {}};
    v8f am[4] = {{}, {}, {}, {}};

    for (int kt = 0; kt < KT_E; ++kt) {
        v16bf bg = *reinterpret_cast<const v16bf*>(wg + ((kt * NT + wave) << 9) + lane * 16);
        v16bf bm = *reinterpret_cast<const v16bf*>(wm + ((kt * NT + wave) << 9) + lane * 16);
#pragma unroll
        for (int r = 0; r < 4; ++r) {
            const unsigned short* rp = zt + (r * 16 + row) * ZP + kt * 32 + hi * 8;
            v16bf a = cat8(*reinterpret_cast<const v8bf*>(rp),
                           *reinterpret_cast<const v8bf*>(rp + 16));
            ag[r] = __builtin_amdgcn_wmma_f32_16x16x32_bf16(false, a, false, bg, (short)0, ag[r], false, false);
            am[r] = __builtin_amdgcn_wmma_f32_16x16x32_bf16(false, a, false, bm, (short)0, am[r], false, false);
        }
    }

    const int c = wave * 16 + row;                 // output channel for this lane
    const float bgv = gb[c], bmv = mb[c];
#pragma unroll
    for (int r = 0; r < 4; ++r) {
#pragma unroll
        for (int v = 0; v < 8; ++v) {
            int m = r * 16 + v + hi * 8;
            int d = dsts[m];
            float g = sigm(ag[r][v] + bgv);
            float s = am[r][v] + bmv;
            s = s * sigm(s);
            atomicAdd(&agg[d * HH + c], g * s);
        }
    }
}

// ---------------- BN stats over N (per channel) of agg/deg ----------------
__global__ void k_bnstats(const float* agg, const float* deg, float* mean, float* inv) {
    int c = blockIdx.x, t = threadIdx.x;
    float s = 0.f, s2 = 0.f;
    for (int n = t; n < NN; n += 256) {
        float d = deg[n]; d = d < 1.f ? 1.f : d;
        float x = agg[n * HH + c] / d;
        s += x; s2 += x * x;
    }
    __shared__ float sh[256], sh2[256];
    sh[t] = s; sh2[t] = s2; __syncthreads();
    for (int o = 128; o > 0; o >>= 1) {
        if (t < o) { sh[t] += sh[t + o]; sh2[t] += sh2[t + o]; }
        __syncthreads();
    }
    if (t == 0) {
        float m = sh[0] / (float)NN;
        float v = sh2[0] / (float)NN - m * m;
        mean[c] = m; inv[c] = rsqrtf(v + EPSV);
    }
}

// ---------------- h = silu(h + BN(agg/deg)) ----------------
__global__ void k_update(const float* agg, const float* deg, const float* mean, const float* inv,
                         const float* g, const float* b, float* h,
                         unsigned short* h_bf, unsigned short* states, int slot) {
    int idx = blockIdx.x * 256 + threadIdx.x;
    if (idx >= NN * HH) return;
    int n = idx >> 7, c = idx & 127;
    float d = deg[n]; d = d < 1.f ? 1.f : d;
    float hn = (agg[idx] / d - mean[c]) * inv[c] * g[c] + b[c];
    float x = h[idx] + hn;
    x = x * sigm(x);
    h[idx] = x;
    unsigned short bb = f2b(x);
    h_bf[idx] = bb;
    states[n * 512 + slot * HH + c] = bb;
}

// ---------------- JK: jk_out = states(N x 512) @ jk_w + jk_b via WMMA ----------------
__global__ void __launch_bounds__(128)
k_jk(const unsigned short* __restrict__ states, const unsigned short* __restrict__ wj,
     const float* __restrict__ jb, float* __restrict__ out) {
    const int t0 = blockIdx.x * 16;
    const int tid = threadIdx.x;
    const int lane = tid & 31, wave = tid >> 5;
    const int n0 = wave * 2, n1 = n0 + 1;
    const int row = lane & 15, hi = lane >> 4;
    v8f a0 = {}, a1 = {};
    for (int kt = 0; kt < KT_J; ++kt) {
        const unsigned short* rp = states + (t0 + row) * 512 + kt * 32 + hi * 8;
        v16bf a = cat8(*reinterpret_cast<const v8bf*>(rp),
                       *reinterpret_cast<const v8bf*>(rp + 16));
        v16bf b0 = *reinterpret_cast<const v16bf*>(wj + ((kt * NT + n0) << 9) + lane * 16);
        v16bf b1 = *reinterpret_cast<const v16bf*>(wj + ((kt * NT + n1) << 9) + lane * 16);
        a0 = __builtin_amdgcn_wmma_f32_16x16x32_bf16(false, a, false, b0, (short)0, a0, false, false);
        a1 = __builtin_amdgcn_wmma_f32_16x16x32_bf16(false, a, false, b1, (short)0, a1, false, false);
    }
    const int c0 = n0 * 16 + row, c1 = n1 * 16 + row;
    const float jb0 = jb[c0], jb1 = jb[c1];
#pragma unroll
    for (int v = 0; v < 8; ++v) {
        int node = t0 + v + hi * 8;
        out[node * HH + c0] = a0[v] + jb0;
        out[node * HH + c1] = a1[v] + jb1;
    }
}

// ---------------- pooling ----------------
__global__ void k_cnt(const int* batch, float* cnt) {
    int n = blockIdx.x * 256 + threadIdx.x;
    if (n < NN) atomicAdd(&cnt[batch[n]], 1.0f);
}
__global__ void k_pool(const float* jk, const int* batch, float* psum, unsigned* pmax) {
    int idx = blockIdx.x * 256 + threadIdx.x;
    if (idx >= NN * HH) return;
    int n = idx >> 7, c = idx & 127;
    int b = batch[n];
    float v = jk[idx];
    atomicAdd(&psum[b * HH + c], v);
    atomicMax(&pmax[b * HH + c], fenc(v));
}

// ---------------- fc1 over [mean | max] ----------------
__global__ void k_fc1(const float* psum, const unsigned* pmax, const float* cnt,
                      const float* w, const float* bias, float* x1) {
    int idx = blockIdx.x * 256 + threadIdx.x;
    if (idx >= BB * HH) return;
    int b = idx >> 7, c = idx & 127;
    float cn = cnt[b];
    float cl = cn < 1.f ? 1.f : cn;
    float acc = bias[c];
    for (int k = 0; k < HH; ++k)
        acc = fmaf(psum[b * HH + k] / cl, w[k * HH + c], acc);
    for (int k = 0; k < HH; ++k) {
        float mx = (cn > 0.f) ? fdec(pmax[b * HH + k]) : 0.f;
        acc = fmaf(mx, w[(HH + k) * HH + c], acc);
    }
    x1[idx] = acc;
}

__global__ void k_fcbn(const float* x1, float* mean, float* inv) {
    int c = blockIdx.x, t = threadIdx.x;
    float s = 0.f, s2 = 0.f;
    for (int b = t; b < BB; b += 256) {
        float x = x1[b * HH + c];
        s += x; s2 += x * x;
    }
    __shared__ float sh[256], sh2[256];
    sh[t] = s; sh2[t] = s2; __syncthreads();
    for (int o = 128; o > 0; o >>= 1) {
        if (t < o) { sh[t] += sh[t + o]; sh2[t] += sh2[t + o]; }
        __syncthreads();
    }
    if (t == 0) {
        float m = sh[0] / (float)BB;
        float v = sh2[0] / (float)BB - m * m;
        mean[c] = m; inv[c] = rsqrtf(v + EPSV);
    }
}

__global__ void k_final(const float* x1, const float* mean, const float* inv,
                        const float* g, const float* b,
                        const float* w2, const float* b2, float* out) {
    int bi = blockIdx.x, c = threadIdx.x;
    float v = (x1[bi * HH + c] - mean[c]) * inv[c] * g[c] + b[c];
    v = v * sigm(v);
    __shared__ float sh[128];
    sh[c] = v * w2[c];
    __syncthreads();
    for (int o = 64; o > 0; o >>= 1) {
        if (c < o) sh[c] += sh[c + o];
        __syncthreads();
    }
    if (c == 0) out[bi] = sh[0] + b2[0];
}

// ---------------- host launcher ----------------
extern "C" void kernel_launch(void* const* d_in, const int* in_sizes, int n_in,
                              void* d_out, int out_size, void* d_ws, size_t ws_size,
                              hipStream_t stream) {
    (void)in_sizes; (void)n_in; (void)out_size; (void)ws_size;
    const float* nf     = (const float*)d_in[0];
    const int*   eidx   = (const int*)  d_in[1];
    const float* ef     = (const float*)d_in[2];
    const int*   batch  = (const int*)  d_in[3];
    const float* emb_w  = (const float*)d_in[4];
    const float* emb_b  = (const float*)d_in[5];
    const float* gate_w = (const float*)d_in[6];
    const float* gate_b = (const float*)d_in[7];
    const float* msg_w  = (const float*)d_in[8];
    const float* msg_b  = (const float*)d_in[9];
    const float* bn_g   = (const float*)d_in[10];
    const float* bn_b   = (const float*)d_in[11];
    const float* jk_w   = (const float*)d_in[12];
    const float* jk_b   = (const float*)d_in[13];
    const float* fc1_w  = (const float*)d_in[14];
    const float* fc1_b  = (const float*)d_in[15];
    const float* fcbn_g = (const float*)d_in[16];
    const float* fcbn_b = (const float*)d_in[17];
    const float* fc2_w  = (const float*)d_in[18];
    const float* fc2_b  = (const float*)d_in[19];
    float* out = (float*)d_out;

    // ---- carve workspace ----
    char* ws = (char*)d_ws;
    size_t off = 0;
    auto carve = [&](size_t bytes) -> char* {
        char* p = ws + off;
        off = (off + bytes + 255) & ~(size_t)255;
        return p;
    };
    float*          deg     = (float*)         carve((size_t)NN * 4);
    float*          h       = (float*)         carve((size_t)NN * HH * 4);
    unsigned short* h_bf    = (unsigned short*)carve((size_t)NN * HH * 2);
    unsigned short* states  = (unsigned short*)carve((size_t)NN * 512 * 2);
    float*          agg     = (float*)         carve((size_t)NN * HH * 4);
    unsigned short* ef_bf   = (unsigned short*)carve((size_t)EE * EDIM * 2);
    unsigned short* wg_frag = (unsigned short*)carve((size_t)3 * KT_E * NT * 512 * 2);
    unsigned short* wm_frag = (unsigned short*)carve((size_t)3 * KT_E * NT * 512 * 2);
    unsigned short* wj_frag = (unsigned short*)carve((size_t)KT_J * NT * 512 * 2);
    float*          jk_out  = (float*)         carve((size_t)NN * HH * 4);
    float*          bn_m    = (float*)         carve(HH * 4);
    float*          bn_i    = (float*)         carve(HH * 4);
    float*          psum    = (float*)         carve((size_t)BB * HH * 4);
    unsigned*       pmax    = (unsigned*)      carve((size_t)BB * HH * 4);
    float*          cnt     = (float*)         carve(BB * 4);
    float*          x1      = (float*)         carve((size_t)BB * HH * 4);
    float*          fm      = (float*)         carve(HH * 4);
    float*          fi      = (float*)         carve(HH * 4);

    auto cdiv = [](int a, int b) { return (a + b - 1) / b; };

    // ---- prep: conversions, fragment weights, embedding, degrees ----
    k_cvt_ef<<<cdiv(EE * EDIM, 256), 256, 0, stream>>>(ef, ef_bf, EE * EDIM);
    for (int l = 0; l < 3; ++l) {
        k_format_w<<<cdiv(KT_E * NT * 512, 256), 256, 0, stream>>>(
            gate_w + (size_t)l * ZDIM * HH, wg_frag + (size_t)l * KT_E * NT * 512, ZDIM, KT_E);
        k_format_w<<<cdiv(KT_E * NT * 512, 256), 256, 0, stream>>>(
            msg_w + (size_t)l * ZDIM * HH, wm_frag + (size_t)l * KT_E * NT * 512, ZDIM, KT_E);
    }
    k_format_w<<<cdiv(KT_J * NT * 512, 256), 256, 0, stream>>>(jk_w, wj_frag, 512, KT_J);
    k_embed<<<cdiv(NN * HH, 256), 256, 0, stream>>>(nf, emb_w, emb_b, h, h_bf, states);
    k_fillf<<<cdiv(NN, 256), 256, 0, stream>>>(deg, 0.f, NN);
    k_deg<<<cdiv(EE, 256), 256, 0, stream>>>(eidx, deg);

    // ---- 3 gated conv layers ----
    for (int l = 0; l < 3; ++l) {
        k_fillf<<<cdiv(NN * HH, 256), 256, 0, stream>>>(agg, 0.f, NN * HH);
        k_edge<<<EE / TE, 256, 0, stream>>>(
            h_bf, ef_bf, eidx,
            wg_frag + (size_t)l * KT_E * NT * 512,
            wm_frag + (size_t)l * KT_E * NT * 512,
            gate_b + (size_t)l * HH, msg_b + (size_t)l * HH, agg);
        k_bnstats<<<HH, 256, 0, stream>>>(agg, deg, bn_m, bn_i);
        k_update<<<cdiv(NN * HH, 256), 256, 0, stream>>>(
            agg, deg, bn_m, bn_i, bn_g + (size_t)l * HH, bn_b + (size_t)l * HH,
            h, h_bf, states, l + 1);
    }

    // ---- JK projection ----
    k_jk<<<NN / 16, 128, 0, stream>>>(states, wj_frag, jk_b, jk_out);

    // ---- pooling ----
    k_fillf<<<cdiv(BB * HH, 256), 256, 0, stream>>>(psum, 0.f, BB * HH);
    k_fillu<<<cdiv(BB * HH, 256), 256, 0, stream>>>(pmax, 0x007fffffu, BB * HH); // enc(-inf)
    k_fillf<<<cdiv(BB, 256), 256, 0, stream>>>(cnt, 0.f, BB);
    k_cnt<<<cdiv(NN, 256), 256, 0, stream>>>(batch, cnt);
    k_pool<<<cdiv(NN * HH, 256), 256, 0, stream>>>(jk_out, batch, psum, pmax);

    // ---- head ----
    k_fc1<<<cdiv(BB * HH, 256), 256, 0, stream>>>(psum, pmax, cnt, fc1_w, fc1_b, x1);
    k_fcbn<<<HH, 256, 0, stream>>>(x1, fm, fi);
    k_final<<<BB, 128, 0, stream>>>(x1, fm, fi, fcbn_g, fcbn_b, fc2_w, fc2_b, out);
}
